// BridgeLayer2_xb_72000831750146
// MI455X (gfx1250) — matrix operand order repaired and checked by reading, not compiled, not run on hardware
//
#include <hip/hip_runtime.h>
#include <hip/hip_bf16.h>
#include <math.h>

// ---------------------------------------------------------------------------
// CDNA5 / gfx1250 VMamba bridge layer.
// GEMM: v_wmma_f32_16x16x32_f16, operands pre-packed as fp16 so each lane's
// fragment is two aligned b128 loads (A row-major M x Kp, B transposed N x Kp).
// Scan: one thread per (b,k,d) channel, h[16] state in registers, v_exp_f32.
// ---------------------------------------------------------------------------

typedef __attribute__((ext_vector_type(16))) _Float16 v16h;
typedef __attribute__((ext_vector_type(8)))  _Float16 v8h;
typedef __attribute__((ext_vector_type(8)))  float    v8f;

#define GEMM_WAVES 8

// C[M,N] (+)= A[M,Kp] * Bt[N,Kp]^T. A, Bt fp16, K pre-padded (Kp % 32 == 0),
// Bt pre-padded to Np = ntN*16 rows. One wave per 16x16 tile; no divergence
// except the store-side n<N guard.
__global__ void k_wmma_gemm16(const _Float16* __restrict__ A, const _Float16* __restrict__ Bt,
                              float* __restrict__ C, long long rsC, long long csC,
                              int N, int Kp, int ntN, int tiles, int beta)
{
    int wid  = threadIdx.x >> 5;
    int lane = threadIdx.x & 31;
    int tile = blockIdx.x * GEMM_WAVES + wid;
    if (tile >= tiles) return;                 // wave-uniform
    int tm  = (tile / ntN) * 16;
    int tn  = (tile % ntN) * 16;
    int grp = lane >> 4;
    int l16 = lane & 15;

    const _Float16* Ap = A  + (size_t)(tm + l16) * Kp + grp * 8;   // 16B aligned
    const _Float16* Bp = Bt + (size_t)(tn + l16) * Kp + grp * 8;   // 16B aligned

    v8f acc = {};
    for (int kk = 0; kk < Kp; kk += 32) {
        v8h a0 = *(const v8h*)(Ap + kk);        // K = grp*8 + 0..7
        v8h a1 = *(const v8h*)(Ap + kk + 16);   // K = 16 + grp*8 + 0..7
        v8h b0 = *(const v8h*)(Bp + kk);        // K = grp*8 + 0..7
        v8h b1 = *(const v8h*)(Bp + kk + 16);   // K = 16 + grp*8 + 0..7
        v16h a = __builtin_shufflevector(a0, a1, 0,1,2,3,4,5,6,7,8,9,10,11,12,13,14,15);
        v16h b = __builtin_shufflevector(b0, b1, 0,1,2,3,4,5,6,7,8,9,10,11,12,13,14,15);
        acc = __builtin_amdgcn_wmma_f32_16x16x32_f16(false, a, false, b, (short)0, acc, false, false);
    }

    int ncol = tn + l16;
    if (ncol < N) {
#pragma unroll
        for (int r = 0; r < 8; ++r) {
            int m = tm + r + grp * 8;          // D: VGPR r -> M=r (grp0) / r+8 (grp1)
            long long off = (long long)m * rsC + (long long)ncol * csC;
            float v = acc[r];
            if (beta) v += C[off];
            C[off] = v;
        }
    }
}

// Weight prep: dst[n*Kp + k] = fp16(src[k*srs + n*scs]), zero-padded.
__global__ void k_prep_w(const float* __restrict__ src, long long srs, long long scs,
                         _Float16* __restrict__ dst, int K, int N, int Kp, int Np)
{
    long long i = (long long)blockIdx.x * blockDim.x + threadIdx.x;
    long long tot = (long long)Np * Kp;
    if (i >= tot) return;
    int k = (int)(i % Kp);
    int n = (int)(i / Kp);
    float v = (k < K && n < N) ? src[(long long)k * srs + (long long)n * scs] : 0.0f;
    dst[i] = (_Float16)v;
}

// (B,C,L) channel-major -> fp16 (B*L, Kp) row-major, K-padded with zeros.
__global__ void k_chw_to_lc16(const float* __restrict__ in, _Float16* __restrict__ out,
                              int B, int Cch, int L, int Kp)
{
    long long i = (long long)blockIdx.x * blockDim.x + threadIdx.x;
    long long tot = (long long)B * L * Kp;
    if (i >= tot) return;
    int c = (int)(i % Kp);
    long long t = i / Kp;
    int l = (int)(t % L);
    int b = (int)(t / L);
    float v = (c < Cch) ? in[((size_t)b * Cch + c) * (size_t)L + l] : 0.0f;
    out[i] = (_Float16)v;
}

__device__ __forceinline__ float silu_f(float x) { return x / (1.0f + __expf(-x)); }

// Depthwise 3x3 SAME conv + SiLU. Input xz rows (B*HW, 2*Di); out (B, Di, H, W).
__global__ void k_conv2d_silu(const float* __restrict__ xz, const float* __restrict__ cw,
                              const float* __restrict__ cb, float* __restrict__ xc,
                              int B, int Di, int H, int W, int twoDi)
{
    long long i = (long long)blockIdx.x * blockDim.x + threadIdx.x;
    long long tot = (long long)B * Di * H * W;
    if (i >= tot) return;
    int w = (int)(i % W);
    long long t = i / W;
    int h = (int)(t % H); t /= H;
    int d = (int)(t % Di);
    int b = (int)(t / Di);
    int L = H * W;
    float acc = cb[d];
#pragma unroll
    for (int dy = 0; dy < 3; ++dy) {
        int hh = h + dy - 1;
        if (hh < 0 || hh >= H) continue;
#pragma unroll
        for (int dx = 0; dx < 3; ++dx) {
            int ww = w + dx - 1;
            if (ww < 0 || ww >= W) continue;
            acc += cw[d * 9 + dy * 3 + dx] *
                   xz[((size_t)b * L + (size_t)hh * W + ww) * twoDi + d];
        }
    }
    xc[(((size_t)b * Di) + d) * (size_t)L + (size_t)h * W + w] = silu_f(acc);
}

// Depthwise k=3 SAME 1D conv + SiLU. Out (B, Di, L).
__global__ void k_conv1d_silu(const float* __restrict__ xz, const float* __restrict__ cw,
                              const float* __restrict__ cb, float* __restrict__ xc,
                              int B, int Di, int L, int twoDi)
{
    long long i = (long long)blockIdx.x * blockDim.x + threadIdx.x;
    long long tot = (long long)B * Di * L;
    if (i >= tot) return;
    int l = (int)(i % L);
    long long t = i / L;
    int d = (int)(t % Di);
    int b = (int)(t / Di);
    float acc = cb[d];
#pragma unroll
    for (int dx = 0; dx < 3; ++dx) {
        int ll = l + dx - 1;
        if (ll < 0 || ll >= L) continue;
        acc += cw[d * 3 + dx] * xz[((size_t)b * L + ll) * twoDi + d];
    }
    xc[(((size_t)b * Di) + d) * (size_t)L + l] = silu_f(acc);
}

// xs (f32, d-major, for scan) and xsT (fp16, l-major rows of Di, GEMM A operand)
__global__ void k_build_xs_2d(const float* __restrict__ xc, float* __restrict__ xs,
                              _Float16* __restrict__ xsT, int B, int Di, int H, int W)
{
    long long i = (long long)blockIdx.x * blockDim.x + threadIdx.x;
    int L = H * W;
    long long tot = (long long)B * Di * L;
    if (i >= tot) return;
    int l = (int)(i % L);
    long long t = i / L;
    int d = (int)(t % Di);
    int b = (int)(t / Di);
    float v = xc[(((size_t)b * Di) + d) * (size_t)L + l];
    _Float16 vh = (_Float16)v;
    int h = l / W, w = l % W;
    int lw = w * H + h;
    int pos[4] = { l, lw, L - 1 - l, L - 1 - lw };
#pragma unroll
    for (int k = 0; k < 4; ++k) {
        xs [(((size_t)(b * 4 + k) * Di) + d) * (size_t)L + pos[k]]  = v;
        xsT[(((size_t)(b * 4 + k) * L) + pos[k]) * (size_t)Di + d]  = vh;
    }
}

__global__ void k_build_xs_1d(const float* __restrict__ xc, float* __restrict__ xs,
                              _Float16* __restrict__ xsT, int B, int Di, int L)
{
    long long i = (long long)blockIdx.x * blockDim.x + threadIdx.x;
    long long tot = (long long)B * Di * L;
    if (i >= tot) return;
    int l = (int)(i % L);
    long long t = i / L;
    int d = (int)(t % Di);
    int b = (int)(t / Di);
    float v = xc[(((size_t)b * Di) + d) * (size_t)L + l];
    _Float16 vh = (_Float16)v;
    int pos[2] = { l, L - 1 - l };
#pragma unroll
    for (int k = 0; k < 2; ++k) {
        xs [(((size_t)(b * 2 + k) * Di) + d) * (size_t)L + pos[k]] = v;
        xsT[(((size_t)(b * 2 + k) * L) + pos[k]) * (size_t)Di + d] = vh;
    }
}

// Selective scan with fused dt projection + softplus.
// xdbl rows (L, Cc): [0,dtr)=dt_raw, [dtr,dtr+16)=B_t, [dtr+16,dtr+32)=C_t.
__global__ void k_scan(const float* __restrict__ xs, const float* __restrict__ xdbl,
                       const float* __restrict__ dt_w, const float* __restrict__ dt_b,
                       const float* __restrict__ A_logs, const float* __restrict__ Ds,
                       float* __restrict__ ys,
                       int K, int Di, int L, int Cc, int dtr, int nth)
{
    int t = blockIdx.x * blockDim.x + threadIdx.x;
    if (t >= nth) return;
    int d = t % Di;
    int k = (t / Di) % K;
    int b = t / (Di * K);

    const float* rows = xdbl + ((size_t)(b * K) + k) * (size_t)L * Cc;
    const float* xp   = xs   + (((size_t)(b * K) + k) * Di + d) * (size_t)L;
    float*       yp   = ys   + (((size_t)(b * K) + k) * Di + d) * (size_t)L;

    float dtw[6];
    for (int r = 0; r < dtr; ++r) dtw[r] = dt_w[((size_t)k * Di + d) * dtr + r];
    float dtb = dt_b[(size_t)k * Di + d];
    float Dv  = Ds  [(size_t)k * Di + d];
    float Av[16];
#pragma unroll
    for (int n = 0; n < 16; ++n) Av[n] = -__expf(A_logs[((size_t)k * Di + d) * 16 + n]);

    float h[16];
#pragma unroll
    for (int n = 0; n < 16; ++n) h[n] = 0.0f;

    for (int l = 0; l < L; ++l) {
        const float* row = rows + (size_t)l * Cc;
        float s = dtb;
        for (int r = 0; r < dtr; ++r) s += dtw[r] * row[r];
        float dt = (s > 30.0f) ? s : log1pf(__expf(s));   // softplus
        float xv = xp[l];
        float dx = dt * xv;
        float y = 0.0f;
#pragma unroll
        for (int n = 0; n < 16; ++n) {
            float hn = __expf(dt * Av[n]) * h[n] + dx * row[dtr + n];
            h[n] = hn;
            y += hn * row[dtr + 16 + n];
        }
        yp[l] = y + Dv * xv;
    }
}

// Combine 4 directions -> (B*L, Di) f32 row-major
__global__ void k_combine_2d(const float* __restrict__ ys, float* __restrict__ yc,
                             int B, int Di, int H, int W)
{
    long long i = (long long)blockIdx.x * blockDim.x + threadIdx.x;
    int L = H * W;
    long long tot = (long long)B * Di * L;
    if (i >= tot) return;
    int l = (int)(i % L);
    long long t = i / L;
    int d = (int)(t % Di);
    int b = (int)(t / Di);
    int h = l / W, w = l % W;
    int lw = w * H + h;
    const float* y0 = ys + (((size_t)(b * 4 + 0) * Di) + d) * (size_t)L;
    const float* y1 = ys + (((size_t)(b * 4 + 1) * Di) + d) * (size_t)L;
    const float* y2 = ys + (((size_t)(b * 4 + 2) * Di) + d) * (size_t)L;
    const float* y3 = ys + (((size_t)(b * 4 + 3) * Di) + d) * (size_t)L;
    float s = y0[l] + y1[lw] + y2[L - 1 - l] + y3[L - 1 - lw];
    yc[((size_t)b * L + l) * Di + d] = s;
}

__global__ void k_combine_1d(const float* __restrict__ ys, float* __restrict__ yc,
                             int B, int Di, int L)
{
    long long i = (long long)blockIdx.x * blockDim.x + threadIdx.x;
    long long tot = (long long)B * Di * L;
    if (i >= tot) return;
    int l = (int)(i % L);
    long long t = i / L;
    int d = (int)(t % Di);
    int b = (int)(t / Di);
    const float* y0 = ys + (((size_t)(b * 2 + 0) * Di) + d) * (size_t)L;
    const float* y1 = ys + (((size_t)(b * 2 + 1) * Di) + d) * (size_t)L;
    yc[((size_t)b * L + l) * Di + d] = y0[l] + y1[L - 1 - l];
}

// layernorm(y) * silu(z) -> fp16 GEMM A operand (rows of Di, Di % 32 == 0)
__global__ void k_ln_gate16(const float* __restrict__ y, const float* __restrict__ xz,
                            const float* __restrict__ sc, const float* __restrict__ bi,
                            _Float16* __restrict__ g, int rows, int Di, int twoDi)
{
    int r = blockIdx.x * blockDim.x + threadIdx.x;
    if (r >= rows) return;
    const float* yr = y + (size_t)r * Di;
    const float* zr = xz + (size_t)r * twoDi + Di;
    _Float16* gr = g + (size_t)r * Di;
    float m = 0.0f;
    for (int d = 0; d < Di; ++d) m += yr[d];
    m /= (float)Di;
    float v = 0.0f;
    for (int d = 0; d < Di; ++d) { float t = yr[d] - m; v += t * t; }
    v /= (float)Di;
    float rs = rsqrtf(v + 1e-5f);
    for (int d = 0; d < Di; ++d) {
        float t = (yr[d] - m) * rs * sc[d] + bi[d];
        gr[d] = (_Float16)(t * silu_f(zr[d]));
    }
}

// layernorm -> fp16 padded (B*L, Kp) GEMM A operand
__global__ void k_ln16(const float* __restrict__ x, _Float16* __restrict__ o,
                       const float* __restrict__ sc, const float* __restrict__ bi,
                       int rows, int C, int Kp)
{
    int r = blockIdx.x * blockDim.x + threadIdx.x;
    if (r >= rows) return;
    const float* xr = x + (size_t)r * C;
    _Float16* orow = o + (size_t)r * Kp;
    float m = 0.0f;
    for (int d = 0; d < C; ++d) m += xr[d];
    m /= (float)C;
    float v = 0.0f;
    for (int d = 0; d < C; ++d) { float t = xr[d] - m; v += t * t; }
    v /= (float)C;
    float rs = rsqrtf(v + 1e-5f);
    for (int d = 0; d < C; ++d) orow[d] = (_Float16)((xr[d] - m) * rs * sc[d] + bi[d]);
    for (int d = C; d < Kp; ++d) orow[d] = (_Float16)0.0f;
}

// F (per-batch 903168 floats: 602112 of o1 then 301056 of o2) -> d_out
__global__ void k_final(const float* __restrict__ F, float* __restrict__ out, int total)
{
    int i = blockIdx.x * blockDim.x + threadIdx.x;
    if (i >= total) return;
    const int O1B = 602112, O2B = 301056, FB = 903168, O1T = 2 * 602112;
    if (i < O1T) {
        int b = i / O1B, r = i % O1B;
        out[i] = F[(size_t)b * FB + r];
    } else {
        int j = i - O1T;
        int b = j / O2B, r = j % O2B;
        out[i] = F[(size_t)b * FB + O1B + r];
    }
}

// ---------------------------------------------------------------------------
// Host orchestration
// ---------------------------------------------------------------------------

struct SSParams {
    const float *in_w, *conv_w, *conv_b, *xp_w, *dt_w, *dt_b, *A_logs, *Ds, *ln_s, *ln_b, *out_w;
};

static inline int nb(long long n) { return (int)((n + 255) / 256); }

static void gemm16(hipStream_t stream, const _Float16* A, const _Float16* Bt,
                   float* C, long long rsC, long long csC, int M, int N, int Kp, int beta)
{
    int ntN = (N + 15) / 16;
    int tiles = (M / 16) * ntN;
    int blocks = (tiles + GEMM_WAVES - 1) / GEMM_WAVES;
    k_wmma_gemm16<<<blocks, GEMM_WAVES * 32, 0, stream>>>(A, Bt, C, rsC, csC, N, Kp, ntN, tiles, beta);
}

static SSParams load_params(void* const* d_in, int base)
{
    SSParams p;
    p.in_w   = (const float*)d_in[base + 0];
    p.conv_w = (const float*)d_in[base + 1];
    p.conv_b = (const float*)d_in[base + 2];
    p.xp_w   = (const float*)d_in[base + 3];
    p.dt_w   = (const float*)d_in[base + 4];
    p.dt_b   = (const float*)d_in[base + 5];
    p.A_logs = (const float*)d_in[base + 6];
    p.Ds     = (const float*)d_in[base + 7];
    p.ln_s   = (const float*)d_in[base + 8];
    p.ln_b   = (const float*)d_in[base + 9];
    p.out_w  = (const float*)d_in[base + 10];
    return p;
}

// Simple bump allocator over the float workspace arena, 32B-aligned chunks.
struct Arena {
    float* base;
    size_t off;  // in floats
    float* f(size_t n)      { float* p = base + off; off += (n + 7) & ~(size_t)7; return p; }
    _Float16* h(size_t nh)  { _Float16* p = (_Float16*)(base + off); off += (((nh + 1) / 2) + 7) & ~(size_t)7; return p; }
};

// SS2D: input (B, dm, H, W); output written (dm, HW)-major into F + b*fb + fo.
static void run_ss2d(hipStream_t stream, const float* x_chw, const SSParams& p,
                     int B, int dm, int Di, int H, int W, int dtr,
                     float* arena_base, float* F, long long fb, long long fo)
{
    const int K = 4, L = H * W, BL = B * L, Cc = dtr + 32, twoDi = 2 * Di;
    const int KpIn = (dm + 31) & ~31;
    const int NpXp = 48;  // (Cc+15)&~15 for Cc=35/38
    Arena ar{arena_base, 0};
    float*     xz    = ar.f((size_t)BL * twoDi);
    float*     xc    = ar.f((size_t)B * Di * L);
    float*     xs    = ar.f((size_t)B * K * Di * L);
    float*     xdbl  = ar.f((size_t)B * K * L * Cc);
    float*     ys    = ar.f((size_t)B * K * Di * L);
    float*     ycomb = ar.f((size_t)BL * Di);
    _Float16*  a16   = ar.h((size_t)BL * KpIn);
    _Float16*  xsT   = ar.h((size_t)B * K * L * Di);
    _Float16*  g16   = ar.h((size_t)BL * Di);
    _Float16*  wtin  = ar.h((size_t)twoDi * KpIn);
    _Float16*  wtxp  = ar.h((size_t)K * NpXp * Di);
    _Float16*  wtout = ar.h((size_t)dm * Di);

    // Weight packing (fp16, transposed, padded)
    k_prep_w<<<nb((long long)twoDi * KpIn), 256, 0, stream>>>(p.in_w, twoDi, 1, wtin, dm, twoDi, KpIn, twoDi);
    for (int k = 0; k < K; ++k)
        k_prep_w<<<nb((long long)NpXp * Di), 256, 0, stream>>>(p.xp_w + (size_t)k * Cc * Di, 1, Di,
                                                               wtxp + (size_t)k * NpXp * Di, Di, Cc, Di, NpXp);
    k_prep_w<<<nb((long long)dm * Di), 256, 0, stream>>>(p.out_w, dm, 1, wtout, Di, dm, Di, dm);

    // in_proj
    k_chw_to_lc16<<<nb((long long)BL * KpIn), 256, 0, stream>>>(x_chw, a16, B, dm, L, KpIn);
    gemm16(stream, a16, wtin, xz, twoDi, 1, BL, twoDi, KpIn, 0);
    // conv + silu, directional sequences
    k_conv2d_silu<<<nb((long long)B * Di * L), 256, 0, stream>>>(xz, p.conv_w, p.conv_b, xc, B, Di, H, W, twoDi);
    k_build_xs_2d<<<nb((long long)B * Di * L), 256, 0, stream>>>(xc, xs, xsT, B, Di, H, W);
    // x_proj per (b, k)
    for (int b = 0; b < B; ++b)
        for (int k = 0; k < K; ++k)
            gemm16(stream, xsT + ((size_t)(b * K + k) * L) * Di, wtxp + (size_t)k * NpXp * Di,
                   xdbl + (size_t)(b * K + k) * L * Cc, Cc, 1, L, Cc, Di, 0);
    // selective scan
    int nth = B * K * Di;
    k_scan<<<nb(nth), 256, 0, stream>>>(xs, xdbl, p.dt_w, p.dt_b, p.A_logs, p.Ds, ys, K, Di, L, Cc, dtr, nth);
    // combine + layernorm*silu(z) -> fp16
    k_combine_2d<<<nb((long long)B * Di * L), 256, 0, stream>>>(ys, ycomb, B, Di, H, W);
    k_ln_gate16<<<nb(BL), 256, 0, stream>>>(ycomb, xz, p.ln_s, p.ln_b, g16, BL, Di, twoDi);
    // out_proj -> (dm, HW)-major region of F
    for (int b = 0; b < B; ++b)
        gemm16(stream, g16 + (size_t)b * L * Di, wtout, F + (size_t)b * fb + fo, 1, L, L, dm, Di, 0);
}

// SS1D: input Fn16 (B*Lf, KpIn) fp16; result ACCUMULATED into F (row-major f32).
static void run_ss1d(hipStream_t stream, const _Float16* Fn16, const SSParams& p,
                     int B, int dm, int Di, int Lf, int dtr,
                     float* arena_base, float* F)
{
    const int K = 2, BL = B * Lf, Cc = dtr + 32, twoDi = 2 * Di;
    const int KpIn = (dm + 31) & ~31;
    const int NpXp = 48;
    Arena ar{arena_base, 0};
    float*     xz    = ar.f((size_t)BL * twoDi);
    float*     xc    = ar.f((size_t)B * Di * Lf);
    float*     xs    = ar.f((size_t)B * K * Di * Lf);
    float*     xdbl  = ar.f((size_t)B * K * Lf * Cc);
    float*     ys    = ar.f((size_t)B * K * Di * Lf);
    float*     ycomb = ar.f((size_t)BL * Di);
    _Float16*  xsT   = ar.h((size_t)B * K * Lf * Di);
    _Float16*  g16   = ar.h((size_t)BL * Di);
    _Float16*  wtin  = ar.h((size_t)twoDi * KpIn);
    _Float16*  wtxp  = ar.h((size_t)K * NpXp * Di);
    _Float16*  wtout = ar.h((size_t)dm * Di);

    k_prep_w<<<nb((long long)twoDi * KpIn), 256, 0, stream>>>(p.in_w, twoDi, 1, wtin, dm, twoDi, KpIn, twoDi);
    for (int k = 0; k < K; ++k)
        k_prep_w<<<nb((long long)NpXp * Di), 256, 0, stream>>>(p.xp_w + (size_t)k * Cc * Di, 1, Di,
                                                               wtxp + (size_t)k * NpXp * Di, Di, Cc, Di, NpXp);
    k_prep_w<<<nb((long long)dm * Di), 256, 0, stream>>>(p.out_w, dm, 1, wtout, Di, dm, Di, dm);

    gemm16(stream, Fn16, wtin, xz, twoDi, 1, BL, twoDi, KpIn, 0);
    k_conv1d_silu<<<nb((long long)B * Di * Lf), 256, 0, stream>>>(xz, p.conv_w, p.conv_b, xc, B, Di, Lf, twoDi);
    k_build_xs_1d<<<nb((long long)B * Di * Lf), 256, 0, stream>>>(xc, xs, xsT, B, Di, Lf);
    for (int b = 0; b < B; ++b)
        for (int k = 0; k < K; ++k)
            gemm16(stream, xsT + ((size_t)(b * K + k) * Lf) * Di, wtxp + (size_t)k * NpXp * Di,
                   xdbl + (size_t)(b * K + k) * Lf * Cc, Cc, 1, Lf, Cc, Di, 0);
    int nth = B * K * Di;
    k_scan<<<nb(nth), 256, 0, stream>>>(xs, xdbl, p.dt_w, p.dt_b, p.A_logs, p.Ds, ys, K, Di, Lf, Cc, dtr, nth);
    k_combine_1d<<<nb((long long)B * Di * Lf), 256, 0, stream>>>(ys, ycomb, B, Di, Lf);
    k_ln_gate16<<<nb(BL), 256, 0, stream>>>(ycomb, xz, p.ln_s, p.ln_b, g16, BL, Di, twoDi);
    // residual: F += g16 @ out_w (single contiguous launch)
    gemm16(stream, g16, wtout, F, dm, 1, BL, dm, Di, 1);
}

extern "C" void kernel_launch(void* const* d_in, const int* in_sizes, int n_in,
                              void* d_out, int out_size, void* d_ws, size_t ws_size,
                              hipStream_t stream)
{
    (void)in_sizes; (void)n_in; (void)ws_size;

    const float* C1 = (const float*)d_in[0];   // (2, 48, 112, 112)
    const float* C2 = (const float*)d_in[1];   // (2, 96, 56, 56)
    SSParams p1 = load_params(d_in, 2);
    SSParams p2 = load_params(d_in, 13);
    SSParams pf = load_params(d_in, 24);
    const float* n1s = (const float*)d_in[35];
    const float* n1b = (const float*)d_in[36];

    const int B = 2;
    const int Lf = 18816;                       // 12544 + 6272 rows of F
    const long long FB = (long long)Lf * 48;    // 903168 floats per batch

    float*     ws    = (float*)d_ws;
    float*     F     = ws;                      // (B, 18816, 48) f32
    _Float16*  Fn16  = (_Float16*)(ws + 1806336);  // (B*18816, 64) fp16, padded
    float*     arena = ws + 1806336 + 1204224;

    // Stage 1: ss2d(C1) -> F1 region ((48, 12544)-major per batch)
    run_ss2d(stream, C1, p1, B, 48, 96, 112, 112, 3, arena, F, FB, 0);
    // Stage 2: ss2d(C2) -> F2 region ((96, 3136)-major per batch)
    run_ss2d(stream, C2, p2, B, 96, 192, 56, 56, 6, arena, F, FB, 602112);
    // Stage 3: Fn16 = fp16(layernorm(F)), K-padded to 64
    k_ln16<<<nb((long long)B * Lf), 256, 0, stream>>>(F, Fn16, n1s, n1b, B * Lf, 48, 64);
    // Stage 4: F += ss1d(Fn16)
    run_ss1d(stream, Fn16, pf, B, 48, 96, Lf, 3, arena, F);
    // Stage 5: emit (o1, o2)
    k_final<<<nb(out_size), 256, 0, stream>>>(F, (float*)d_out, out_size);
}
